// EncoderLayer_39144331936384
// MI455X (gfx1250) — compile-verified, matrix operand order checked by reading, not run on hardware
//
#include <hip/hip_runtime.h>
#include <hip/hip_bf16.h>
#include <cstdint>

// ---- problem constants (match reference) ----
#define B_   2
#define L_   2048
#define D_   1024
#define H_   16
#define DH_  64
#define FF_  4096
#define EPS_ 1e-5f
#define NEGINF (-3.402823466e38f)

typedef __attribute__((ext_vector_type(16))) __bf16 v16bf;
typedef __attribute__((ext_vector_type(8)))  __bf16 bf16x8;
typedef __attribute__((ext_vector_type(4)))  __bf16 bf16x4;
typedef __attribute__((ext_vector_type(8)))  float  v8f;
typedef int v4i_ __attribute__((vector_size(16)));   // matches async-LDS builtin param type

// ---- CDNA5 async global->LDS staging (ASYNCcnt path), with safe fallback ----
#if defined(__AMDGCN__) && __has_builtin(__builtin_amdgcn_global_load_async_to_lds_b128) && \
    __has_builtin(__builtin_amdgcn_s_wait_asynccnt)
#define USE_ASYNC_LDS 1
#else
#define USE_ASYNC_LDS 0
#endif

__device__ __forceinline__ void stage_b128(const __bf16* gsrc, __bf16* ldst) {
#if USE_ASYNC_LDS
    // GLOBAL_LOAD_ASYNC_TO_LDS_B128: 16B per lane, tracked by ASYNCcnt
    __builtin_amdgcn_global_load_async_to_lds_b128(
        (__attribute__((address_space(1))) v4i_*)gsrc,
        (__attribute__((address_space(3))) v4i_*)ldst, 0, 0);
#else
    *(bf16x8*)ldst = *(const bf16x8*)gsrc;
#endif
}

// ------------------ block reduction helpers (blockDim == 256) ------------------
__device__ __forceinline__ float blk_sum(float v, float* red) {
    const int t = threadIdx.x;
    red[t] = v; __syncthreads();
    for (int s = 128; s > 0; s >>= 1) { if (t < s) red[t] += red[t + s]; __syncthreads(); }
    float r = red[0]; __syncthreads(); return r;
}
__device__ __forceinline__ float blk_max(float v, float* red) {
    const int t = threadIdx.x;
    red[t] = v; __syncthreads();
    for (int s = 128; s > 0; s >>= 1) { if (t < s) red[t] = fmaxf(red[t], red[t + s]); __syncthreads(); }
    float r = red[0]; __syncthreads(); return r;
}

// ------------------ elementwise f32 -> bf16 cast (vectorized x4) ------------------
__global__ void cast_f32_bf16(const float* __restrict__ in, __bf16* __restrict__ out, int n4) {
    int i = blockIdx.x * blockDim.x + threadIdx.x;
    if (i < n4) {
        const float4 v = ((const float4*)in)[i];
        bf16x4 o; o[0] = (__bf16)v.x; o[1] = (__bf16)v.y; o[2] = (__bf16)v.z; o[3] = (__bf16)v.w;
        ((bf16x4*)out)[i] = o;
    }
}

// ------------------ transpose + cast: out[n][k] = (bf16) in[k][n] ------------------
__global__ __launch_bounds__(256) void transpose_cvt(const float* __restrict__ in,
                                                     __bf16* __restrict__ out,
                                                     int K, int N) {
    __shared__ float tile[32][33];
    const int n0 = blockIdx.x * 32, k0 = blockIdx.y * 32;
    const int tx = threadIdx.x & 31, ty = threadIdx.x >> 5;   // 32 x 8
    for (int i = 0; i < 32; i += 8) tile[ty + i][tx] = in[(size_t)(k0 + ty + i) * N + n0 + tx];
    __syncthreads();
    for (int i = 0; i < 32; i += 8)
        out[(size_t)(n0 + ty + i) * K + k0 + tx] = (__bf16)tile[tx][ty + i];
}

// ------------------ bf16 WMMA GEMM: C = A(MxK) * Bt(NxK)^T + bias, opt ReLU ------------------
// Block tile 256x128 (M x N), 256 threads = 8 waves in 4x2; wave tile 64x64 = 4x4 WMMA tiles.
// 16 WMMAs per K-step vs 16 ds_load_b128 (8 fragments) per wave. Double-buffered LDS (60KB),
// async global->LDS staging (6 x B128/thread/stage) overlapped with compute.
// Requires: M % 256 == 0, N % 128 == 0, K % 32 == 0.
__global__ __launch_bounds__(256) void gemm_bf16_wmma(
    const __bf16* __restrict__ A,
    const __bf16* __restrict__ Bt,
    const float*  __restrict__ bias,
    float*        __restrict__ Cf,     // optional f32 output
    __bf16*       __restrict__ Cb,     // optional bf16 output
    int M, int N, int K, int relu)
{
    __shared__ __align__(16) __bf16 As[2][256 * 40];   // 2 x (256 rows x (32+8 pad) halves)
    __shared__ __align__(16) __bf16 Bs[2][128 * 40];

    const int tid  = threadIdx.x;
    const int wave = tid >> 5;
    const int lane = tid & 31;
    const int ll   = lane & 15;
    const int hi   = lane >> 4;
    const int wm   = wave & 3;      // 4 row slots of 64
    const int wn   = wave >> 2;     // 2 col slots of 64
    const int brow = blockIdx.y * 256;
    const int bcol = blockIdx.x * 128;

    v8f acc[4][4] = {};

    // staging: A tile 256x32 -> 1024 chunks (4/thread); B tile 128x32 -> 512 chunks (2/thread)
    int arw[4], akk[4], brw[2], bkk[2];
    const __bf16* gA[4]; const __bf16* gB[2];
#pragma unroll
    for (int i = 0; i < 4; i++) {
        const int c = tid + i * 256; arw[i] = c >> 2; akk[i] = (c & 3) * 8;
        gA[i] = &A[(size_t)(brow + arw[i]) * K + akk[i]];
    }
#pragma unroll
    for (int i = 0; i < 2; i++) {
        const int c = tid + i * 256; brw[i] = c >> 2; bkk[i] = (c & 3) * 8;
        gB[i] = &Bt[(size_t)(bcol + brw[i]) * K + bkk[i]];
    }

    auto stage = [&](int buf, int kb) {
#pragma unroll
        for (int i = 0; i < 4; i++) stage_b128(gA[i] + kb, &As[buf][arw[i] * 40 + akk[i]]);
#pragma unroll
        for (int i = 0; i < 2; i++) stage_b128(gB[i] + kb, &Bs[buf][brw[i] * 40 + bkk[i]]);
#if !USE_ASYNC_LDS
        if (kb + 32 < K) {
            __builtin_prefetch(gA[0] + kb + 32, 0, 1);
            __builtin_prefetch(gB[0] + kb + 32, 0, 1);
        }
#endif
    };

    stage(0, 0);                       // prologue: fill buffer 0
    int buf = 0;
    for (int kb = 0; kb < K; kb += 32, buf ^= 1) {
        if (kb + 32 < K) {
            stage(buf ^ 1, kb + 32);   // kick off next tile into the other buffer
#if USE_ASYNC_LDS
            // in-order completion: <=6 outstanding means the current tile's 6 are done
            __builtin_amdgcn_s_wait_asynccnt(6);
#endif
        } else {
#if USE_ASYNC_LDS
            __builtin_amdgcn_s_wait_asynccnt(0);
#endif
        }
        __syncthreads();               // all waves' staging of buf complete

        // ISA 16-bit A layout: lanes 0-15 -> K 0..7 & 16..23 ; lanes 16-31 -> K 8..15 & 24..31
        v16bf af[4], bfr[4];
#pragma unroll
        for (int mt = 0; mt < 4; ++mt) {
            const __bf16* rp = &As[buf][(wm * 64 + mt * 16 + ll) * 40];
            bf16x8 lo = *(const bf16x8*)(rp + hi * 8);
            bf16x8 hh = *(const bf16x8*)(rp + 16 + hi * 8);
            af[mt] = __builtin_shufflevector(lo, hh, 0,1,2,3,4,5,6,7,8,9,10,11,12,13,14,15);
        }
#pragma unroll
        for (int nt = 0; nt < 4; ++nt) {
            const __bf16* rp = &Bs[buf][(wn * 64 + nt * 16 + ll) * 40];
            bf16x8 lo = *(const bf16x8*)(rp + hi * 8);
            bf16x8 hh = *(const bf16x8*)(rp + 16 + hi * 8);
            bfr[nt] = __builtin_shufflevector(lo, hh, 0,1,2,3,4,5,6,7,8,9,10,11,12,13,14,15);
        }
#pragma unroll
        for (int mt = 0; mt < 4; ++mt)
#pragma unroll
            for (int nt = 0; nt < 4; ++nt)
                acc[mt][nt] = __builtin_amdgcn_wmma_f32_16x16x32_bf16(
                    false, af[mt], false, bfr[nt], (short)0, acc[mt][nt], false, false);
        __syncthreads();               // done reading buf; it may be overwritten next iter
    }

    // epilogue: bias (+ReLU), f32 and/or bf16 stores
#pragma unroll
    for (int mt = 0; mt < 4; ++mt) {
#pragma unroll
        for (int nt = 0; nt < 4; ++nt) {
            const int col  = bcol + wn * 64 + nt * 16 + ll;
            const float bv = bias ? bias[col] : 0.f;
            const int rbase = brow + wm * 64 + mt * 16 + hi * 8;
#pragma unroll
            for (int r = 0; r < 8; ++r) {
                float v = acc[mt][nt][r] + bv;
                if (relu) v = fmaxf(v, 0.f);
                const size_t o = (size_t)(rbase + r) * N + col;
                if (Cf) Cf[o] = v;
                if (Cb) Cb[o] = (__bf16)v;
            }
        }
    }
}

// ------------------ sampled sparsity measure: M = max_s(QKs) - sum_s(QKs)/L ------------------
// Q/K rows live in the fused QKV buffer: row stride ld (=3*D), head offset h*64.
__global__ void probe_scores(const float* __restrict__ Q, const float* __restrict__ Km,
                             const int* __restrict__ idx, float* __restrict__ Msc,
                             int S, int ld) {
    int gid = blockIdx.x * blockDim.x + threadIdx.x;
    if (gid >= B_ * H_ * L_) return;
    const int l = gid % L_, h = (gid / L_) % H_, b = gid / (H_ * L_);
    const float4* q4 = (const float4*)&Q[((size_t)(b * L_ + l)) * ld + h * DH_];
    float4 q[16];
#pragma unroll
    for (int i = 0; i < 16; i++) q[i] = q4[i];
    float mx = NEGINF, sm = 0.f;
    for (int s = 0; s < S; s++) {
        const int ls = idx[l * S + s];
        const float4* k4 = (const float4*)&Km[((size_t)(b * L_ + ls)) * ld + h * DH_];
        float acc = 0.f;
#pragma unroll
        for (int i = 0; i < 16; i++) {
            float4 kv = k4[i];
            acc += q[i].x * kv.x + q[i].y * kv.y + q[i].z * kv.z + q[i].w * kv.w;
        }
        mx = fmaxf(mx, acc); sm += acc;
    }
    Msc[gid] = mx - sm * (1.f / (float)L_);
}

// ------------------ iterative top-S selection per (b,h) ------------------
__global__ __launch_bounds__(256) void topk_sel(const float* __restrict__ Msc,
                                                int* __restrict__ Mtop, int S) {
    __shared__ float vals[L_];
    __shared__ float redv[256];
    __shared__ int   redi[256];
    const int bh = blockIdx.x, tid = threadIdx.x;
    for (int i = tid; i < L_; i += 256) vals[i] = Msc[(size_t)bh * L_ + i];
    __syncthreads();
    for (int t = 0; t < S; t++) {
        float bvv = NEGINF; int bii = 0;
        for (int i = tid; i < L_; i += 256) { float v = vals[i]; if (v > bvv) { bvv = v; bii = i; } }
        redv[tid] = bvv; redi[tid] = bii; __syncthreads();
        for (int s2 = 128; s2 > 0; s2 >>= 1) {
            if (tid < s2) {
                if (redv[tid + s2] > redv[tid]) { redv[tid] = redv[tid + s2]; redi[tid] = redi[tid + s2]; }
            }
            __syncthreads();
        }
        if (tid == 0) { Mtop[bh * S + t] = redi[0]; vals[redi[0]] = NEGINF; }
        __syncthreads();
    }
}

// ------------------ masked softmax(Q_red K^T)*V for one selected query ------------------
__global__ __launch_bounds__(256) void reduced_attn(
    const float* __restrict__ Q, const float* __restrict__ Km, const float* __restrict__ V,
    const int* __restrict__ Mtop, float* __restrict__ upd, int S, int ld)
{
    __shared__ __align__(16) float qsh[DH_];
    __shared__ float p[L_];
    __shared__ float red[256];
    const int u  = blockIdx.x % S;
    const int bh = blockIdx.x / S;
    const int h = bh % H_, b = bh / H_;
    const int tid = threadIdx.x;
    const int lsel = Mtop[bh * S + u];
    if (tid < DH_) qsh[tid] = Q[((size_t)(b * L_ + lsel)) * ld + h * DH_ + tid];
    __syncthreads();
    const float4* q4 = (const float4*)qsh;
    float lmax = NEGINF;
    for (int l = tid; l < L_; l += 256) {
        float sc = NEGINF;
        if (l <= lsel) {                               // causal mask: l > lsel -> -inf
            const float4* k4 = (const float4*)&Km[((size_t)(b * L_ + l)) * ld + h * DH_];
            float acc = 0.f;
#pragma unroll
            for (int i = 0; i < 16; i++) {
                float4 kv = k4[i]; float4 qv = q4[i];
                acc += qv.x * kv.x + qv.y * kv.y + qv.z * kv.z + qv.w * kv.w;
            }
            sc = acc * 0.125f;                          // 1/sqrt(64)
            lmax = fmaxf(lmax, sc);
        }
        p[l] = sc;
    }
    const float m = blk_max(lmax, red);
    float lsum = 0.f;
    for (int l = tid; l < L_; l += 256) {
        float e = (l <= lsel) ? __expf(p[l] - m) : 0.f;
        p[l] = e; lsum += e;
    }
    const float Z = blk_sum(lsum, red);
    // upd[d] = sum_l p[l] * V[l][d] / Z  (64 lanes per d-group, 4 l-groups)
    const int d = tid & 63, g = tid >> 6;
    float acc = 0.f;
    const float* Vb = &V[(size_t)(b * L_) * ld + h * DH_ + d];
    for (int l = g; l < L_; l += 4) acc += p[l] * Vb[(size_t)l * ld];
    red[tid] = acc; __syncthreads();
    if (g == 0) {
        float tot = red[d] + red[64 + d] + red[128 + d] + red[192 + d];
        upd[(size_t)blockIdx.x * DH_ + d] = tot / Z;
    }
}

// ------------------ cumsum of V along l (chunked 2-phase scan); ctx has stride D ------------------
__global__ void cumsum_local(const float* __restrict__ V, float* __restrict__ ctx,
                             float* __restrict__ csum, int ld) {
    int t = blockIdx.x * blockDim.x + threadIdx.x;
    if (t >= B_ * H_ * DH_ * 8) return;
    const int d = t & 63, chunk = (t >> 6) & 7, bh = t >> 9;
    const int h = bh % H_, b = bh / H_;
    const size_t col = h * DH_ + d;
    const size_t rowbase = (size_t)b * L_ + chunk * 256;
    float run = 0.f;
    for (int i = 0; i < 256; i++) {
        run += V[(rowbase + i) * ld + col];
        ctx[(rowbase + i) * D_ + col] = run;
    }
    csum[t] = run;
}
__global__ void cumsum_fix(float* __restrict__ ctx, const float* __restrict__ csum) {
    int t = blockIdx.x * blockDim.x + threadIdx.x;
    if (t >= B_ * H_ * DH_ * 8) return;
    const int d = t & 63, chunk = (t >> 6) & 7, bh = t >> 9;
    if (chunk == 0) return;
    const int h = bh % H_, b = bh / H_;
    float offs = 0.f;
    for (int c = 0; c < chunk; c++) offs += csum[(bh << 9) + (c << 6) + d];
    const size_t col = h * DH_ + d;
    const size_t rowbase = (size_t)b * L_ + chunk * 256;
    for (int i = 0; i < 256; i++) ctx[(rowbase + i) * D_ + col] += offs;
}

// ------------------ scatter: ctx[b,h,Mtop[u],:] = upd[b,h,u,:] ------------------
__global__ void scatter_upd(float* __restrict__ ctx, const float* __restrict__ upd,
                            const int* __restrict__ Mtop, int S) {
    int t = blockIdx.x * blockDim.x + threadIdx.x;
    if (t >= B_ * H_ * S * DH_) return;
    const int d = t & 63, u = (t >> 6) % S, bh = t / (S * 64);
    const int h = bh % H_, b = bh / H_;
    const int lsel = Mtop[bh * S + u];
    ctx[((size_t)(b * L_ + lsel)) * D_ + h * DH_ + d] = upd[(size_t)(bh * S + u) * DH_ + d];
}

// ------------------ fused residual + LayerNorm, emits f32 and optional bf16 ------------------
__global__ __launch_bounds__(256) void ln_add(const float* __restrict__ X, const float* __restrict__ Y,
                                              const float* __restrict__ g, const float* __restrict__ bb,
                                              float* __restrict__ outF, __bf16* __restrict__ outB) {
    __shared__ float red[256];
    const int row = blockIdx.x, tid = threadIdx.x;
    float s[4]; float lsum = 0.f;
#pragma unroll
    for (int i = 0; i < 4; i++) {
        const int c = tid + i * 256;
        const float v = X[(size_t)row * D_ + c] + Y[(size_t)row * D_ + c];
        s[i] = v; lsum += v;
    }
    const float mu = blk_sum(lsum, red) * (1.f / D_);
    float lvar = 0.f;
#pragma unroll
    for (int i = 0; i < 4; i++) { const float dv = s[i] - mu; lvar += dv * dv; }
    const float var = blk_sum(lvar, red) * (1.f / D_);
    const float inv = rsqrtf(var + EPS_);
#pragma unroll
    for (int i = 0; i < 4; i++) {
        const int c = tid + i * 256;
        const float o = (s[i] - mu) * inv * g[c] + bb[c];
        if (outF) outF[(size_t)row * D_ + c] = o;
        if (outB) outB[(size_t)row * D_ + c] = (__bf16)o;
    }
}

// =====================================================================================
extern "C" void kernel_launch(void* const* d_in, const int* in_sizes, int n_in,
                              void* d_out, int out_size, void* d_ws, size_t ws_size,
                              hipStream_t stream) {
    (void)n_in; (void)out_size; (void)ws_size;
    const float* x   = (const float*)d_in[0];
    const float* Wq  = (const float*)d_in[1];
    const float* bq  = (const float*)d_in[2];
    const float* Wk  = (const float*)d_in[3];
    const float* bk  = (const float*)d_in[4];
    const float* Wv  = (const float*)d_in[5];
    const float* bv  = (const float*)d_in[6];
    const float* Wo  = (const float*)d_in[7];
    const float* bo  = (const float*)d_in[8];
    const float* g1  = (const float*)d_in[9];
    const float* be1 = (const float*)d_in[10];
    const float* W1  = (const float*)d_in[11];
    const float* b1  = (const float*)d_in[12];
    const float* W2  = (const float*)d_in[13];
    const float* b2  = (const float*)d_in[14];
    const float* g2  = (const float*)d_in[15];
    const float* be2 = (const float*)d_in[16];
    const int*   idx = (const int*)d_in[17];
    const int S = in_sizes[17] / L_;   // sample_k == n_top == 40

    // ---- workspace carve-up (~137 MiB with reuse) ----
    char* base = (char*)d_ws; size_t off = 0;
    auto alloc = [&](size_t bytes) -> void* {
        off = (off + 255) & ~(size_t)255; void* p = base + off; off += bytes; return p;
    };
    const size_t ND = (size_t)B_ * L_ * D_;               // 4M elements
    __bf16* xb   = (__bf16*)alloc(ND * 2);
    // WqT/WkT/WvT allocated back-to-back (2 MiB each, 256-aligned) => contiguous
    // 3072x1024 bf16 matrix = fused QKV weight, transposed
    __bf16* WqT  = (__bf16*)alloc((size_t)D_ * D_ * 2);
    __bf16* WkT  = (__bf16*)alloc((size_t)D_ * D_ * 2);
    __bf16* WvT  = (__bf16*)alloc((size_t)D_ * D_ * 2);
    __bf16* WoT  = (__bf16*)alloc((size_t)D_ * D_ * 2);
    __bf16* W1T  = (__bf16*)alloc((size_t)D_ * FF_ * 2);
    __bf16* W2T  = (__bf16*)alloc((size_t)FF_ * D_ * 2);
    float*  QKV  = (float*) alloc(ND * 3 * 4);            // fused [Q|K|V], row stride 3*D
    float*  ctx  = (float*) alloc(ND * 4);
    __bf16* ctxb = (__bf16*)alloc(ND * 2);
    __bf16* hb   = (__bf16*)alloc((size_t)B_ * L_ * FF_ * 2);
    float*  Msc  = (float*) alloc((size_t)B_ * H_ * L_ * 4);
    int*    Mtop = (int*)   alloc((size_t)B_ * H_ * S * 4);
    float*  upd  = (float*) alloc((size_t)B_ * H_ * S * DH_ * 4);
    float*  csum = (float*) alloc((size_t)B_ * H_ * DH_ * 8 * 4);
    float*  bqkv = (float*) alloc((size_t)3 * D_ * 4);    // concatenated QKV bias
    // QKV buffer is dead after cumsum_local; reuse it for attn_out / x1 / ff
    float*  Qp = QKV, *Kp = QKV + D_, *Vp = QKV + 2 * D_; // head views, stride 3*D
    float*  attn_out = QKV;
    float*  x1f = QKV + ND;
    float*  ff  = QKV + 2 * ND;
    __bf16* x1b = (__bf16*)ctx;                           // ctx f32 dead after ctxb cast

    const int M  = B_ * L_;            // 4096 rows
    const int n4 = (int)(ND / 4);
    const int LDQ = 3 * D_;            // fused QKV row stride

    // 1) bf16 copies (activations) + transposed bf16 weights + fused bias
    cast_f32_bf16<<<(n4 + 255) / 256, 256, 0, stream>>>(x, xb, n4);
    transpose_cvt<<<dim3(D_ / 32, D_ / 32), 256, 0, stream>>>(Wq, WqT, D_, D_);
    transpose_cvt<<<dim3(D_ / 32, D_ / 32), 256, 0, stream>>>(Wk, WkT, D_, D_);
    transpose_cvt<<<dim3(D_ / 32, D_ / 32), 256, 0, stream>>>(Wv, WvT, D_, D_);
    transpose_cvt<<<dim3(D_ / 32, D_ / 32), 256, 0, stream>>>(Wo, WoT, D_, D_);
    transpose_cvt<<<dim3(FF_ / 32, D_ / 32), 256, 0, stream>>>(W1, W1T, D_, FF_);
    transpose_cvt<<<dim3(D_ / 32, FF_ / 32), 256, 0, stream>>>(W2, W2T, FF_, D_);
    hipMemcpyAsync(bqkv,          bq, D_ * 4, hipMemcpyDeviceToDevice, stream);
    hipMemcpyAsync(bqkv + D_,     bk, D_ * 4, hipMemcpyDeviceToDevice, stream);
    hipMemcpyAsync(bqkv + 2 * D_, bv, D_ * 4, hipMemcpyDeviceToDevice, stream);

    // 2) fused QKV projection (one WMMA GEMM, N = 3*D)
    gemm_bf16_wmma<<<dim3(3 * D_ / 128, M / 256), 256, 0, stream>>>(
        xb, WqT /* fused 3072x1024 Bt */, bqkv, QKV, nullptr, M, 3 * D_, D_, 0);

    // 3) ProbSparse attention
    probe_scores<<<(B_ * H_ * L_ + 255) / 256, 256, 0, stream>>>(Qp, Kp, idx, Msc, S, LDQ);
    topk_sel<<<B_ * H_, 256, 0, stream>>>(Msc, Mtop, S);
    reduced_attn<<<B_ * H_ * S, 256, 0, stream>>>(Qp, Kp, Vp, Mtop, upd, S, LDQ);
    const int nscan = B_ * H_ * DH_ * 8;
    cumsum_local<<<(nscan + 255) / 256, 256, 0, stream>>>(Vp, ctx, csum, LDQ);
    cumsum_fix  <<<(nscan + 255) / 256, 256, 0, stream>>>(ctx, csum);
    scatter_upd <<<(B_ * H_ * S * DH_ + 255) / 256, 256, 0, stream>>>(ctx, upd, Mtop, S);

    // 4) output projection + LN1 (residual)
    cast_f32_bf16<<<(n4 + 255) / 256, 256, 0, stream>>>(ctx, ctxb, n4);
    gemm_bf16_wmma<<<dim3(D_ / 128, M / 256), 256, 0, stream>>>(
        ctxb, WoT, bo, attn_out, nullptr, M, D_, D_, 0);
    ln_add<<<M, 256, 0, stream>>>(x, attn_out, g1, be1, x1f, x1b);

    // 5) FFN (WMMA, ReLU fused; intermediate kept bf16) + LN2 (residual) -> d_out
    gemm_bf16_wmma<<<dim3(FF_ / 128, M / 256), 256, 0, stream>>>(
        x1b, W1T, b1, nullptr, hb, M, FF_, D_, 1);
    gemm_bf16_wmma<<<dim3(D_ / 128, M / 256), 256, 0, stream>>>(
        hb, W2T, b2, ff, nullptr, M, D_, FF_, 0);
    ln_add<<<M, 256, 0, stream>>>(x1f, ff, g2, be2, (float*)d_out, nullptr);
}
